// PSMMix3dDitEmbedding_40596030881944
// MI455X (gfx1250) — matrix operand order repaired and checked by reading, not compiled, not run on hardware
//
#include <hip/hip_runtime.h>
#include <hip/hip_bf16.h>
#include <math.h>

typedef __bf16 bf16_t;
typedef __attribute__((ext_vector_type(16))) __bf16 v16bf;
typedef __attribute__((ext_vector_type(8)))  __bf16 v8bf;
typedef __attribute__((ext_vector_type(8)))  float  v8f;

// Problem constants (from reference)
constexpr int B_ = 2, L_ = 512, D_ = 1024, K_ = 128, H_ = 32;

// Output layout (flat f32, concatenated in return order)
constexpr size_t X_OFF   = 0;                          // x            (B,L,D)
constexpr size_t PAD_OFF = X_OFF   + (size_t)B_*L_*D_; // padding_mask (B,L)
constexpr size_t TE_OFF  = PAD_OFF + (size_t)B_*L_;    // time_embed   (B,L,D)
constexpr size_t GAB_OFF = TE_OFF  + (size_t)B_*L_*D_; // graph_attn_bias (B,H,L,L)
constexpr size_t PE_OFF  = GAB_OFF + (size_t)B_*H_*L_*L_; // pos_embedding (B,L,D)

constexpr float MN = 3.402823466e38f;           // finfo(f32).min magnitude
constexpr float SCALING = 0.08838834764831845f; // 128^-0.5
constexpr float INV_SQRT2PI = 0.3989422804014327f;
constexpr int HP = 136;                         // LDS pitch for H tile (halves): 272B rows, 16B aligned

// --------------------------------------------------------------------------
// Prep: swizzle proj_w1 (K,K) and proj_w2 (H,K) f32 -> bf16 WMMA B-fragments.
// B fragment layout (16-bit B, 32x16): lane L: n = L%16, kh = L/16, b[e] = B[16*(e/8)+kh*8+e%8, n]
// GEMM1: H[j,k'] = sum_k E[j,k] * W1[k',k]  -> B[k,n] = W1[t*16+n, kstep*32+k]
// frag index = ((t*4+kstep)*32 + lane)*16 + e
// --------------------------------------------------------------------------
__global__ void prep_wfrag(const float* __restrict__ w1, const float* __restrict__ w2,
                           bf16_t* __restrict__ w1f, bf16_t* __restrict__ w2f) {
  int idx = blockIdx.x * blockDim.x + threadIdx.x;
  if (idx < 8 * 4 * 32 * 16) {               // W1: 8 n-tiles x 4 ksteps
    int e = idx & 15, lane = (idx >> 4) & 31, ks = (idx >> 9) & 3, t = idx >> 11;
    int n = lane & 15, kh = lane >> 4;
    int k = ks * 32 + ((e & 8) << 1) + kh * 8 + (e & 7);
    w1f[idx] = (bf16_t)w1[(t * 16 + n) * K_ + k];
  } else if (idx < 8 * 4 * 32 * 16 + 2 * 4 * 32 * 16) {  // W2: 2 n-tiles x 4 ksteps
    int j = idx - 8 * 4 * 32 * 16;
    int e = j & 15, lane = (j >> 4) & 31, ks = (j >> 9) & 3, t2 = j >> 11;
    int n = lane & 15, kh = lane >> 4;
    int k = ks * 32 + ((e & 8) << 1) + kh * 8 + (e & 7);
    w2f[j] = (bf16_t)w2[(t2 * 16 + n) * K_ + k];
  }
}

__device__ __forceinline__ float gaussv(float x, float mean, float inv, float coef) {
  float z = (x - mean) * inv;
  return coef * __expf(-0.5f * z * z);
}

// Exact-GELU via branch-free erf (Abramowitz-Stegun 7.1.26, |err| < 1.5e-7,
// below the bf16 noise of the WMMA inputs). v_rcp + v_exp, no divergence,
// no IEEE-division expansion.
__device__ __forceinline__ float gelu_f(float x) {
  float y = x * 0.7071067811865475f;
  float ay = fabsf(y);
  float t = __builtin_amdgcn_rcpf(1.0f + 0.3275911f * ay);
  float p = t * (0.254829592f +
            t * (-0.284496736f +
            t * (1.421413741f +
            t * (-1.453152027f +
            t * 1.061405429f))));
  float erfa = 1.0f - p * __expf(-ay * ay);
  float erfy = copysignf(erfa, y);
  return 0.5f * x * (1.0f + erfy);
}

// --------------------------------------------------------------------------
// Main fused kernel: one workgroup (128 thr = 4 waves) per (b, i).
// --------------------------------------------------------------------------
__global__ __launch_bounds__(128) void psm_main(
    const int* __restrict__ token_id, const float* __restrict__ pos,
    const unsigned char* __restrict__ adj, const unsigned char* __restrict__ is_periodic,
    const int* __restrict__ nte, const int* __restrict__ time_step,
    const float* __restrict__ embed_w, const float* __restrict__ ts_w,
    const float* __restrict__ pos_emb_w, const float* __restrict__ pos_feat_w,
    const float* __restrict__ gbf_means, const float* __restrict__ gbf_stds,
    const float* __restrict__ gbf_mul, const float* __restrict__ gbf_bias,
    const float* __restrict__ proj_b1, const float* __restrict__ proj_b2,
    const float* __restrict__ pos_proj_w, const float* __restrict__ pos_proj_b,
    const bf16_t* __restrict__ w1f, const bf16_t* __restrict__ w2f,
    float* __restrict__ out) {

  __shared__ __align__(16) float x_lds[L_];      // mul*dist+bias scalars per j
  __shared__ int   flags_lds[L_];                // bit0: pad_j, bit1: adj2
  __shared__ float d_lds[L_];                    // head-summed bias
  __shared__ float attn_lds[L_];                 // softmax workspace
  __shared__ float pos_lds[L_ * 3];
  __shared__ __align__(16) float gm_lds[K_], gi_lds[K_], gc_lds[K_];
  __shared__ float b1_lds[K_], b2_lds[H_];
  __shared__ float v_lds[K_], ef_lds[K_];
  __shared__ float red_lds[128];
  __shared__ float bcast;
  __shared__ __align__(16) bf16_t h_lds[4][16][HP]; // per-wave 16x128 H tile (padded)

  const int bi = blockIdx.x;
  const int b = bi / L_, i = bi % L_;
  const int tid = threadIdx.x;

  const int tok_i = token_id[b * L_ + i];
  const bool pad_i = (tok_i == 0);
  const bool mol_i = (tok_i <= 129) && (is_periodic[b] == 0);
  const bool row_all_adj = !mol_i;

  const float pix = pos[(b * L_ + i) * 3 + 0];
  const float piy = pos[(b * L_ + i) * 3 + 1];
  const float piz = pos[(b * L_ + i) * 3 + 2];

  // ---------------- Phase 1: per-edge scalars, flags, params ----------------
  for (int j = tid; j < L_; j += 128) {
    float pjx = pos[(b * L_ + j) * 3 + 0];
    float pjy = pos[(b * L_ + j) * 3 + 1];
    float pjz = pos[(b * L_ + j) * 3 + 2];
    pos_lds[3 * j + 0] = pjx; pos_lds[3 * j + 1] = pjy; pos_lds[3 * j + 2] = pjz;
    float dx = pix - pjx, dy = piy - pjy, dz = piz - pjz;
    float dist = sqrtf(dx * dx + dy * dy + dz * dz);
    size_t eo = (((size_t)(b * L_ + i)) * L_ + j) * 2;
    int n0 = nte[eo], n1 = nte[eo + 1];
    float mulv = gbf_mul[n0] + gbf_mul[n1];
    float biav = gbf_bias[n0] + gbf_bias[n1];
    x_lds[j] = mulv * dist + biav;
    bool adjv = adj[((size_t)(b * L_ + i)) * L_ + j] != 0;
    bool adj2 = adjv || row_all_adj;
    bool pad_j = (token_id[b * L_ + j] == 0);
    flags_lds[j] = (pad_j ? 1 : 0) | (adj2 ? 2 : 0);
  }
  if (tid < K_) {
    float sd = fabsf(gbf_stds[tid]) + 1e-5f;
    float gi = 1.0f / sd;
    gm_lds[tid] = gbf_means[tid];
    gi_lds[tid] = gi;
    gc_lds[tid] = gi * INV_SQRT2PI;
    b1_lds[tid] = proj_b1[tid];
  }
  if (tid < H_) b2_lds[tid] = proj_b2[tid];

  const int wave = tid >> 5;
  const int lane = tid & 31;
  const int col  = lane & 15;       // A/B/C column role (lane % 16)
  const int hi   = lane >> 4;       // which 16-lane half
  __syncthreads();

  // ---------------- Phase 2: WMMA GEMMs over 16-row j-tiles ----------------
  for (int jt = wave; jt < L_ / 16; jt += 4) {
    const int base = jt * 16;
    const float xm = x_lds[base + col];    // A-matrix row m = lane%16

    // Build all 4 A fragments (16x32 E slabs) up front: 32 VGPRs live.
    v16bf afrag[4];
    for (int ks = 0; ks < 4; ++ks) {
      const int kb = ks * 32 + (hi << 3);  // A layout: e<8 -> k=kb+e, e>=8 -> k=kb+16+(e-8)
      const float4 gm0 = *(const float4*)&gm_lds[kb];
      const float4 gm1 = *(const float4*)&gm_lds[kb + 4];
      const float4 gm2 = *(const float4*)&gm_lds[kb + 16];
      const float4 gm3 = *(const float4*)&gm_lds[kb + 20];
      const float4 gi0 = *(const float4*)&gi_lds[kb];
      const float4 gi1 = *(const float4*)&gi_lds[kb + 4];
      const float4 gi2 = *(const float4*)&gi_lds[kb + 16];
      const float4 gi3 = *(const float4*)&gi_lds[kb + 20];
      const float4 gc0 = *(const float4*)&gc_lds[kb];
      const float4 gc1 = *(const float4*)&gc_lds[kb + 4];
      const float4 gc2 = *(const float4*)&gc_lds[kb + 16];
      const float4 gc3 = *(const float4*)&gc_lds[kb + 20];
      v16bf a;
      a[0]  = (bf16_t)gaussv(xm, gm0.x, gi0.x, gc0.x);
      a[1]  = (bf16_t)gaussv(xm, gm0.y, gi0.y, gc0.y);
      a[2]  = (bf16_t)gaussv(xm, gm0.z, gi0.z, gc0.z);
      a[3]  = (bf16_t)gaussv(xm, gm0.w, gi0.w, gc0.w);
      a[4]  = (bf16_t)gaussv(xm, gm1.x, gi1.x, gc1.x);
      a[5]  = (bf16_t)gaussv(xm, gm1.y, gi1.y, gc1.y);
      a[6]  = (bf16_t)gaussv(xm, gm1.z, gi1.z, gc1.z);
      a[7]  = (bf16_t)gaussv(xm, gm1.w, gi1.w, gc1.w);
      a[8]  = (bf16_t)gaussv(xm, gm2.x, gi2.x, gc2.x);
      a[9]  = (bf16_t)gaussv(xm, gm2.y, gi2.y, gc2.y);
      a[10] = (bf16_t)gaussv(xm, gm2.z, gi2.z, gc2.z);
      a[11] = (bf16_t)gaussv(xm, gm2.w, gi2.w, gc2.w);
      a[12] = (bf16_t)gaussv(xm, gm3.x, gi3.x, gc3.x);
      a[13] = (bf16_t)gaussv(xm, gm3.y, gi3.y, gc3.y);
      a[14] = (bf16_t)gaussv(xm, gm3.z, gi3.z, gc3.z);
      a[15] = (bf16_t)gaussv(xm, gm3.w, gi3.w, gc3.w);
      afrag[ks] = a;
    }

    // GEMM1: flattened 32-step (t x ks) pipeline with explicit B double-buffer
    // so the next fragment load is issued before the current WMMA consumes
    // its operand (branch-free wrap on the final prefetch).
    {
      v16bf bcur = *(const v16bf*)(w1f + ((size_t)lane << 4));
      v8f acc;
      for (int r = 0; r < 8; ++r) acc[r] = 0.0f;
#pragma unroll
      for (int idx = 0; idx < 32; ++idx) {
        const int nxt = (idx + 1) & 31;
        v16bf bnext = *(const v16bf*)(w1f + (((nxt * 32 + lane)) << 4));
        acc = __builtin_amdgcn_wmma_f32_16x16x32_bf16(
            false, afrag[idx & 3], false, bcur, (short)0, acc, false, false);
        bcur = bnext;
        if ((idx & 3) == 3) {          // tile t = idx>>2 complete (compile-time)
          const int t = idx >> 2;
          const float b1v = b1_lds[t * 16 + col];
          for (int r = 0; r < 8; ++r) {
            float hv = gelu_f(acc[r] + b1v);
            h_lds[wave][r + (hi << 3)][t * 16 + col] = (bf16_t)hv; // C layout: M=r+8*hi, N=col
          }
          for (int r = 0; r < 8; ++r) acc[r] = 0.0f;
        }
      }
    }

    // GEMM2: hoist the 4 H A-fragments, then 2 N-tiles with one live acc each.
    v16bf a2frag[4];
    for (int ks = 0; ks < 4; ++ks) {
      const int kb2 = ks * 32 + (hi << 3);
      const bf16_t* hrow = &h_lds[wave][col][0];
      v8bf lo = *(const v8bf*)(hrow + kb2);
      v8bf hh = *(const v8bf*)(hrow + kb2 + 16);
      a2frag[ks] = __builtin_shufflevector(lo, hh, 0, 1, 2, 3, 4, 5, 6, 7,
                                           8, 9, 10, 11, 12, 13, 14, 15);
    }

    const int jrow0 = base + (hi << 3);
    float raw[2][8];
#pragma unroll
    for (int t2 = 0; t2 < 2; ++t2) {
      v8f acc2;
      for (int r = 0; r < 8; ++r) acc2[r] = 0.0f;
      for (int ks = 0; ks < 4; ++ks) {
        v16bf bf = *(const v16bf*)(w2f + (((t2 * 4 + ks) * 32 + lane) << 4));
        acc2 = __builtin_amdgcn_wmma_f32_16x16x32_bf16(
            false, a2frag[ks], false, bf, (short)0, acc2, false, false);
      }
      const int hdx = t2 * 16 + col;
      const float b2v = b2_lds[hdx];
      float ov[8];
      for (int r = 0; r < 8; ++r) {
        float rv = acc2[r] + b2v;
        raw[t2][r] = rv;
        int fl = flags_lds[jrow0 + r];
        ov[r] = pad_i ? 0.0f
                      : (((fl & 1) || !(fl & 2)) ? -__builtin_inff() : rv);
      }
      float* gp = out + GAB_OFF + ((((size_t)(b * H_ + hdx)) * L_ + i) * L_ + jrow0);
      float4 v0 = {ov[0], ov[1], ov[2], ov[3]};
      float4 v1 = {ov[4], ov[5], ov[6], ov[7]};
      *(float4*)gp = v0;
      *(float4*)(gp + 4) = v1;
    }

    // Head-sum d: butterfly across the 16-lane halves (C layout N striping).
    for (int r = 0; r < 8; ++r) {
      float s = raw[0][r] + raw[1][r];
      s += __shfl_xor(s, 1);
      s += __shfl_xor(s, 2);
      s += __shfl_xor(s, 4);
      s += __shfl_xor(s, 8);
      if (col == 0) d_lds[jrow0 + r] = s;
    }
  }
  __syncthreads();

  // ---------------- Phase 3: softmax, ef, matvecs, outputs ----------------
  float lm = -MN;
  for (int j = tid; j < L_; j += 128) {
    int fl = flags_lds[j];
    bool valid = !pad_i && !(fl & 1) && (fl & 2);
    float dv = (valid ? d_lds[j] : -MN) * SCALING;
    attn_lds[j] = dv;
    lm = fmaxf(lm, dv);
  }
  red_lds[tid] = lm;
  __syncthreads();
  if (tid == 0) {
    float m = -MN;
    for (int t = 0; t < 128; ++t) m = fmaxf(m, red_lds[t]);
    bcast = m;
  }
  __syncthreads();
  const float mx = bcast;
  float ls = 0.0f;
  for (int j = tid; j < L_; j += 128) {
    float e = __expf(attn_lds[j] - mx);
    attn_lds[j] = e;
    ls += e;
  }
  __syncthreads();           // guard red_lds reuse
  red_lds[tid] = ls;
  __syncthreads();
  if (tid == 0) {
    float s = 0.0f;
    for (int t = 0; t < 128; ++t) s += red_lds[t];
    bcast = s;
  }
  __syncthreads();
  const float invsum = 1.0f / bcast;

  // ef[k] and v[k] = (attn @ pe)[k]  (pe recomputed from pos, zero where pad_j)
  {
    const int t = tid;                        // 128 threads == K_
    float w0 = pos_emb_w[t * 3 + 0];
    float w1v = pos_emb_w[t * 3 + 1];
    float w2v = pos_emb_w[t * 3 + 2];
    float mean = gm_lds[t], giv = gi_lds[t], gcv = gc_lds[t];
    float efv = 0.0f, vv = 0.0f;
#pragma unroll 2
    for (int j = 0; j < L_; ++j) {
      int fl = flags_lds[j];
      if (!(fl & 1)) {
        float z = (x_lds[j] - mean) * giv;
        efv += gcv * __expf(-0.5f * z * z);
        float pev = pos_lds[3 * j] * w0 + pos_lds[3 * j + 1] * w1v + pos_lds[3 * j + 2] * w2v;
        vv += attn_lds[j] * pev;
      }
    }
    v_lds[t] = vv * invsum;
    ef_lds[t] = efv;
  }
  __syncthreads();

  // Final: pfe = v @ pos_feat_w^T + ef @ pos_proj_w^T + b ; outputs
  const float* embrow = embed_w + (size_t)tok_i * D_;
  const float* tsrow = ts_w + (size_t)time_step[b * L_ + i] * D_;
#pragma unroll 1
  for (int dd = tid; dd < D_; dd += 128) {
    float pf = pos_proj_b[dd];
    const float* wf = pos_feat_w + (size_t)dd * K_;
    const float* wp = pos_proj_w + (size_t)dd * K_;
#pragma unroll 8
    for (int k = 0; k < K_; ++k) pf += v_lds[k] * wf[k] + ef_lds[k] * wp[k];
    float pemb = pad_i ? 0.0f : pf;
    float te = tsrow[dd];
    size_t o = ((size_t)(b * L_ + i)) * D_ + dd;
    out[X_OFF + o] = embrow[dd] + te + pemb;
    out[TE_OFF + o] = te;
    out[PE_OFF + o] = pemb;
  }
  if (tid == 0) out[PAD_OFF + (size_t)(b * L_ + i)] = pad_i ? 1.0f : 0.0f;
}

extern "C" void kernel_launch(void* const* d_in, const int* in_sizes, int n_in,
                              void* d_out, int out_size, void* d_ws, size_t ws_size,
                              hipStream_t stream) {
  (void)in_sizes; (void)n_in; (void)out_size;
  const int*   token_id   = (const int*)d_in[0];
  const float* pos        = (const float*)d_in[1];
  const unsigned char* adj        = (const unsigned char*)d_in[2];
  const unsigned char* is_per     = (const unsigned char*)d_in[3];
  const int*   nte        = (const int*)d_in[4];
  const int*   time_step  = (const int*)d_in[5];
  const float* embed_w    = (const float*)d_in[6];
  const float* ts_w       = (const float*)d_in[7];
  const float* pos_emb_w  = (const float*)d_in[8];
  const float* pos_feat_w = (const float*)d_in[9];
  const float* gbf_means  = (const float*)d_in[10];
  const float* gbf_stds   = (const float*)d_in[11];
  const float* gbf_mul    = (const float*)d_in[12];
  const float* gbf_bias   = (const float*)d_in[13];
  const float* proj_w1    = (const float*)d_in[14];
  const float* proj_b1    = (const float*)d_in[15];
  const float* proj_w2    = (const float*)d_in[16];
  const float* proj_b2    = (const float*)d_in[17];
  const float* pos_proj_w = (const float*)d_in[18];
  const float* pos_proj_b = (const float*)d_in[19];

  if (ws_size < (16384 + 4096) * sizeof(bf16_t)) return;
  bf16_t* w1f = (bf16_t*)d_ws;
  bf16_t* w2f = w1f + 16384;
  float* out = (float*)d_out;

  prep_wfrag<<<(16384 + 4096 + 255) / 256, 256, 0, stream>>>(proj_w1, proj_w2, w1f, w2f);
  psm_main<<<B_ * L_, 128, 0, stream>>>(
      token_id, pos, adj, is_per, nte, time_step, embed_w, ts_w, pos_emb_w,
      pos_feat_w, gbf_means, gbf_stds, gbf_mul, gbf_bias, proj_b1, proj_b2,
      pos_proj_w, pos_proj_b, w1f, w2f, out);
}
